// GDecode_45243185496760
// MI455X (gfx1250) — matrix operand order repaired and checked by reading, not compiled, not run on hardware
//
#include <hip/hip_runtime.h>
#include <hip/hip_bf16.h>
#include <math.h>

// Problem constants (from reference): B=4, N=512 -> M=2048 rows; G=8; K=64; V=32000
#define MT 2048
#define GG 8
#define KK 64
#define VV 32000

typedef __attribute__((ext_vector_type(16))) __bf16 v16bf;
typedef __attribute__((ext_vector_type(8)))  float  v8f;

__device__ __forceinline__ void split2(float x, __bf16& h, __bf16& l) {
    h = (__bf16)x;                 // RNE to bf16 (top 8 mantissa bits)
    l = (__bf16)(x - (float)h);    // residual (next ~8 mantissa bits)
}

// Workspace layout (bytes):
//   aH   : MT*512*2        =  2,097,152   @ 0
//   aL   : MT*512*2        =  2,097,152   @ 2 MiB
//   psiH : VV*KK*2         =  4,096,000   @ 4 MiB
//   psiL : VV*KK*2         =  4,096,000   @ 4 MiB + 4,096,000
#define WS_AH   0
#define WS_AL   (2u * 1024u * 1024u)
#define WS_PSIH (4u * 1024u * 1024u)
#define WS_PSIL (4u * 1024u * 1024u + 4096000u)

// ---------------------------------------------------------------------------
// Prep 1: split phi into bf16 hi/lo, pre-swizzled into A-fragment lane layout.
// Fragment f = mblock*16 + step, step = h*2 + kb (32-wide K chunk).
// This code's 16-bit A 16x32 layout (contiguous k; B uses the same):
//   lanes 0-15 : M = ln, elems 0..15 = K 0..15
//   lanes 16-31: M = ln, elems 0..15 = K 16..31
// Each lane owns 16 contiguous bf16 (32 B) -> GEMM does one raw v16bf load
// per fragment per lane.
// ---------------------------------------------------------------------------
__global__ __launch_bounds__(256)
void prep_a(const float* __restrict__ phi, __bf16* __restrict__ aH,
            __bf16* __restrict__ aL)
{
    const int gwave = (blockIdx.x * 256 + threadIdx.x) >> 5;  // 0 .. 1023
    const int lane  = threadIdx.x & 31;
    const int ln    = lane & 15;
    const bool hiH  = (lane >= 16);

    const int mblock = gwave >> 4;        // 0..63
    const int step   = gwave & 15;        // 0..15  (= h*2 + kb)
    const int kbase  = step * 32;         // == h*64 + kb*32

    const float* row = phi + (size_t)(mblock * 16 + ln) * (GG * KK);
    const int ako = hiH ? 16 : 0;

    __bf16* dH = aH + ((size_t)gwave * 512) + lane * 16;
    __bf16* dL = aL + ((size_t)gwave * 512) + lane * 16;

#pragma unroll
    for (int i = 0; i < 8; ++i) {
        __bf16 h, l;
        split2(row[kbase + ako + i], h, l);
        dH[i] = h; dL[i] = l;
    }
#pragma unroll
    for (int i = 0; i < 8; ++i) {
        __bf16 h, l;
        split2(row[kbase + ako + 8 + i], h, l);
        dH[8 + i] = h; dL[8 + i] = l;
    }
}

// ---------------------------------------------------------------------------
// Prep 2: elementwise split psi [V,K] f32 -> psiH/psiL bf16 (natural row
// layout; each 32-wide K chunk half is a contiguous 32 B span).
// ---------------------------------------------------------------------------
__global__ __launch_bounds__(256)
void prep_psi(const float* __restrict__ psi, __bf16* __restrict__ pH,
              __bf16* __restrict__ pL)
{
    const int i = blockIdx.x * 256 + threadIdx.x;
    if (i < VV * KK) {
        __bf16 h, l;
        split2(psi[i], h, l);
        pH[i] = h; pL[i] = l;
    }
}

// ---------------------------------------------------------------------------
// GEMM: logits[m, v] = sum_h dot(phi[m, h, :], psi[perm[h, v], :])
// Block = 256 threads = 8 waves. WG tile: 64 (M) x 256 (V).
// Wave tile: 32 (M) x 64 (V) -> acc[2][4] 16x16 f32 accumulators.
// Split-bf16: D += aH*bH + aH*bL + aL*bH  (3 WMMAs per 32-wide K step).
// Each B fragment is reused across the two M sub-blocks: per K-step this is
// 24 WMMAs vs 24 b128 loads (~32 FLOP/byte from L2). All B gathers hit L2
// (psi splits are 8 MB total). No conversion VALU in the hot loop; B gather
// offsets are 32-bit so loads can use the SGPR-base + VGPR-offset form.
// ---------------------------------------------------------------------------
__global__ __launch_bounds__(256, 1)
void gdecode_gemm(const __bf16* __restrict__ aH, const __bf16* __restrict__ aL,
                  const __bf16* __restrict__ psiH, const __bf16* __restrict__ psiL,
                  const long long* __restrict__ perm, float* __restrict__ out)
{
    const int tid  = threadIdx.x;
    const int lane = tid & 31;
    const int wave = tid >> 5;
    const int ln   = lane & 15;
    const bool hiH = (lane >= 16);

    const int m0 = blockIdx.y * 64  + (wave & 1) * 32;   // wave's 32-row M block
    const int v0 = blockIdx.x * 256 + (wave >> 1) * 64;  // wave's 64-col V block

    v8f acc[2][4] = {};

    // Per-step A fragment bases for the two 16-row sub-blocks.
    const int mb0 = m0 >> 4;
    const __bf16* aHp0 = aH + ((size_t)mb0 * 16) * 512 + lane * 16;
    const __bf16* aLp0 = aL + ((size_t)mb0 * 16) * 512 + lane * 16;
    const __bf16* aHp1 = aHp0 + (size_t)16 * 512;
    const __bf16* aLp1 = aLp0 + (size_t)16 * 512;

    // B fragment lane layout (16-bit B 32x16): lanes 0-15 hold K 0..15 of
    // column v0+t*16+ln; lanes 16-31 hold K 16..31 of the same column.
    const unsigned bko = hiH ? 32u : 0u;   // byte offset within 32-K chunk

    const char* psiHb = (const char*)psiH;
    const char* psiLb = (const char*)psiL;

    for (int h = 0; h < GG; ++h) {
        // Gather psi row byte-offsets for this lane's 4 columns (fits in u32).
        unsigned bOff[4];
#pragma unroll
        for (int t = 0; t < 4; ++t) {
            const long long pr = perm[(size_t)h * VV + (v0 + t * 16 + ln)];
            bOff[t] = (unsigned)pr * (KK * 2u) + bko;
        }

#pragma unroll
        for (int kb = 0; kb < 2; ++kb) {
            const size_t stepOff = (size_t)(h * 2 + kb) * 512;
            const v16bf aH0 = *(const v16bf*)(aHp0 + stepOff);
            const v16bf aL0 = *(const v16bf*)(aLp0 + stepOff);
            const v16bf aH1 = *(const v16bf*)(aHp1 + stepOff);
            const v16bf aL1 = *(const v16bf*)(aLp1 + stepOff);

#pragma unroll
            for (int t = 0; t < 4; ++t) {
                const unsigned off = bOff[t] + (unsigned)kb * 64u;
                const v16bf bHf = *(const v16bf*)(psiHb + off);
                const v16bf bLf = *(const v16bf*)(psiLb + off);

                v8f c0 = acc[0][t];
                c0 = __builtin_amdgcn_wmma_f32_16x16x32_bf16(false, aH0, false, bHf, (short)0, c0, false, false);
                c0 = __builtin_amdgcn_wmma_f32_16x16x32_bf16(false, aH0, false, bLf, (short)0, c0, false, false);
                c0 = __builtin_amdgcn_wmma_f32_16x16x32_bf16(false, aL0, false, bHf, (short)0, c0, false, false);
                acc[0][t] = c0;

                v8f c1 = acc[1][t];
                c1 = __builtin_amdgcn_wmma_f32_16x16x32_bf16(false, aH1, false, bHf, (short)0, c1, false, false);
                c1 = __builtin_amdgcn_wmma_f32_16x16x32_bf16(false, aH1, false, bLf, (short)0, c1, false, false);
                c1 = __builtin_amdgcn_wmma_f32_16x16x32_bf16(false, aL1, false, bHf, (short)0, c1, false, false);
                acc[1][t] = c1;
            }
        }
    }

    // C/D layout: VGPR r -> M = r (lanes 0-15) or M = 8+r (lanes 16-31), N = lane%16.
    const int mS = m0 + (hiH ? 8 : 0);
#pragma unroll
    for (int mb = 0; mb < 2; ++mb) {
#pragma unroll
        for (int t = 0; t < 4; ++t) {
            const int v = v0 + t * 16 + ln;
#pragma unroll
            for (int r = 0; r < 8; ++r) {
                out[(size_t)(mS + mb * 16 + r) * VV + v] = acc[mb][t][r];
            }
        }
    }
}

// ---------------------------------------------------------------------------
// In-place row softmax over V=32000. One block (256 threads) per row.
// ---------------------------------------------------------------------------
__global__ __launch_bounds__(256, 2)
void gdecode_softmax(float* __restrict__ out)
{
    const int row = blockIdx.x;
    float* p = out + (size_t)row * VV;
    const int lane = threadIdx.x & 31;
    const int wv   = threadIdx.x >> 5;

    __shared__ float redMax[8];
    __shared__ float redSum[8];

    float m = -INFINITY;
    for (int i = threadIdx.x; i < VV; i += 256) m = fmaxf(m, p[i]);
#pragma unroll
    for (int off = 16; off > 0; off >>= 1) m = fmaxf(m, __shfl_xor(m, off, 32));
    if (lane == 0) redMax[wv] = m;
    __syncthreads();
    m = redMax[0];
#pragma unroll
    for (int w = 1; w < 8; ++w) m = fmaxf(m, redMax[w]);

    float s = 0.f;
    for (int i = threadIdx.x; i < VV; i += 256) {
        const float e = __expf(p[i] - m);
        p[i] = e;
        s += e;
    }
#pragma unroll
    for (int off = 16; off > 0; off >>= 1) s += __shfl_xor(s, off, 32);
    if (lane == 0) redSum[wv] = s;
    __syncthreads();
    s = 0.f;
#pragma unroll
    for (int w = 0; w < 8; ++w) s += redSum[w];
    const float inv = __frcp_rn(s);

    for (int i = threadIdx.x; i < VV; i += 256) p[i] *= inv;
}

// ---------------------------------------------------------------------------
extern "C" void kernel_launch(void* const* d_in, const int* in_sizes, int n_in,
                              void* d_out, int out_size, void* d_ws, size_t ws_size,
                              hipStream_t stream) {
    const float*     phi  = (const float*)d_in[0];      // [B,N,G,K] f32
    const float*     psi  = (const float*)d_in[1];      // [V,K]     f32
    const long long* perm = (const long long*)d_in[2];  // [G,V]     i64
    float* out = (float*)d_out;                         // [B,N,V]   f32

    char* ws = (char*)d_ws;
    __bf16* aH   = (__bf16*)(ws + WS_AH);
    __bf16* aL   = (__bf16*)(ws + WS_AL);
    __bf16* psiH = (__bf16*)(ws + WS_PSIH);
    __bf16* psiL = (__bf16*)(ws + WS_PSIL);

    // Prep: split/convert (tiny; ~12.5 MB total workspace, all L2-resident).
    prep_a<<<dim3(MT * 16 / 8 / 32, 1, 1), 256, 0, stream>>>(phi, aH, aL);   // 1024 waves
    prep_psi<<<dim3((VV * KK + 255) / 256, 1, 1), 256, 0, stream>>>(psi, psiH, psiL);

    // Main GEMM: 125 x 32 blocks of 256 threads (wave tile 32M x 64V).
    dim3 gemmGrid(VV / 256, MT / 64, 1);
    gdecode_gemm<<<gemmGrid, 256, 0, stream>>>(aH, aL, psiH, psiL, perm, out);

    // Softmax in place.
    gdecode_softmax<<<dim3(MT, 1, 1), 256, 0, stream>>>(out);
}
